// RebasinNet_28252294873565
// MI455X (gfx1250) — compile-verified, matrix operand order checked by reading, not run on hardware
//
#include <hip/hip_runtime.h>

typedef __attribute__((ext_vector_type(16))) __bf16 v16bf;
typedef __attribute__((ext_vector_type(4)))  __bf16 v4bf;
typedef __attribute__((ext_vector_type(8)))  float  v8f;

#define H_DIM 4096
#define D_IN  1024
#define D_OUT 1024
#define BATCH 256

// ---- CDNA5 async-copy helpers (ASYNCcnt-tracked, LDS-direct) ----
__device__ __forceinline__ uint32_t lds_lo32(const void* p) {
    // flat shared address: addr[31:0] is the wave-relative LDS byte address
    return (uint32_t)(uintptr_t)p;
}
__device__ __forceinline__ void async_ld_b128(uint32_t lds, uint64_t gaddr) {
    asm volatile("global_load_async_to_lds_b128 %0, %1, off"
                 :: "v"(lds), "v"(gaddr) : "memory");
}
__device__ __forceinline__ void wait_async0() {
    asm volatile("s_wait_asynccnt 0x0" ::: "memory");
}

// ======================= Sinkhorn =======================
__global__ __launch_bounds__(256) void sk_row(const float* __restrict__ src,
                                              float* __restrict__ dst) {
    __shared__ float red[8];
    const int tid  = threadIdx.x;
    const int lane = tid & 31, wave = tid >> 5;
    const size_t row = blockIdx.x;
    const float* r = src + row * H_DIM;
    float x[16];
    float m = -__builtin_inff();
#pragma unroll
    for (int i = 0; i < 16; ++i) { x[i] = r[tid + (i << 8)]; m = fmaxf(m, x[i]); }
#pragma unroll
    for (int off = 16; off; off >>= 1) m = fmaxf(m, __shfl_xor(m, off, 32));
    if (lane == 0) red[wave] = m;
    __syncthreads();
    float bm = red[0];
#pragma unroll
    for (int i = 1; i < 8; ++i) bm = fmaxf(bm, red[i]);
    __syncthreads();
    float s = 0.f;
#pragma unroll
    for (int i = 0; i < 16; ++i) s += __expf(x[i] - bm);
#pragma unroll
    for (int off = 16; off; off >>= 1) s += __shfl_xor(s, off, 32);
    if (lane == 0) red[wave] = s;
    __syncthreads();
    float bs = 0.f;
#pragma unroll
    for (int i = 0; i < 8; ++i) bs += red[i];
    const float lse = bm + __logf(bs);
    float* w = dst + row * H_DIM;
#pragma unroll
    for (int i = 0; i < 16; ++i) w[tid + (i << 8)] = x[i] - lse;
}

__global__ __launch_bounds__(256) void sk_col(float* __restrict__ lt) {
    __shared__ float sm[8][32], ss[8][32];
    const int tx = threadIdx.x & 31, ty = threadIdx.x >> 5;
    const size_t c = (size_t)blockIdx.x * 32 + tx;
    float m = -__builtin_inff(), s = 0.f;
    for (int r = ty; r < H_DIM; r += 8) {
        float x = lt[(size_t)r * H_DIM + c];
        if (x > m) { s = s * __expf(m - x) + 1.f; m = x; }
        else         s += __expf(x - m);
    }
    sm[ty][tx] = m; ss[ty][tx] = s;
    __syncthreads();
    if (ty == 0) {
        float M = sm[0][tx], S = ss[0][tx];
#pragma unroll
        for (int i = 1; i < 8; ++i) {
            float m2 = sm[i][tx], s2 = ss[i][tx];
            float nm = fmaxf(M, m2);
            S = S * __expf(M - nm) + s2 * __expf(m2 - nm);
            M = nm;
        }
        sm[0][tx] = M; ss[0][tx] = S;
    }
    __syncthreads();
    const float lse = sm[0][tx] + __logf(ss[0][tx]);
    for (int r = ty; r < H_DIM; r += 8)
        lt[(size_t)r * H_DIM + c] -= lse;
}

__global__ __launch_bounds__(256) void exp_to_bf16(const float* __restrict__ lt,
                                                   __bf16* __restrict__ out) {
    const size_t i = ((size_t)blockIdx.x * 256 + threadIdx.x) * 4;
    const float4 v = *(const float4*)(lt + i);
    v4bf o;
    o.x = (__bf16)__expf(v.x); o.y = (__bf16)__expf(v.y);
    o.z = (__bf16)__expf(v.z); o.w = (__bf16)__expf(v.w);
    *(v4bf*)(out + i) = o;
}

// ======================= conversions =======================
__global__ __launch_bounds__(256) void cvt_to_bf16(const float* __restrict__ src,
                                                   __bf16* __restrict__ dst) {
    const size_t i = ((size_t)blockIdx.x * 256 + threadIdx.x) * 4;
    const float4 v = *(const float4*)(src + i);
    v4bf o;
    o.x = (__bf16)v.x; o.y = (__bf16)v.y; o.z = (__bf16)v.z; o.w = (__bf16)v.w;
    *(v4bf*)(dst + i) = o;
}

__global__ __launch_bounds__(256) void transpose_to_bf16(const float* __restrict__ src,
                                                         __bf16* __restrict__ dst,
                                                         int rows, int cols) {
    __shared__ float tile[32][33];
    const int tx = threadIdx.x & 31, ty = threadIdx.x >> 5;
    const int c0 = blockIdx.x * 32, r0 = blockIdx.y * 32;
    for (int i = ty; i < 32; i += 8)
        tile[i][tx] = src[(size_t)(r0 + i) * cols + c0 + tx];
    __syncthreads();
    for (int i = ty; i < 32; i += 8)
        dst[(size_t)(c0 + i) * rows + r0 + tx] = (__bf16)tile[tx][i];
}

__global__ __launch_bounds__(256) void gemv_bf16(const __bf16* __restrict__ P,
                                                 const float* __restrict__ b,
                                                 float* __restrict__ out) {
    const int lane = threadIdx.x & 31;
    const size_t row = (size_t)blockIdx.x * 8 + (threadIdx.x >> 5);
    const __bf16* pr = P + row * H_DIM;
    float s = 0.f;
    for (int j = lane; j < H_DIM; j += 32) s += (float)pr[j] * b[j];
#pragma unroll
    for (int off = 16; off; off >>= 1) s += __shfl_xor(s, off, 32);
    if (lane == 0) out[row] = s;
}

// ======================= WMMA GEMM (NT) =======================
// C(M,N) = A(MxK) * Bt(NxK)^T ; bf16 in, fp32 accum.
// 128x256 block tile, 8 waves in 2(M)x4(N), each wave 64x64 via 4x4 WMMA tiles.
// Double-buffered LDS staged with GLOBAL_LOAD_ASYNC_TO_LDS_B128 (ASYNCcnt).
#define BM 128
#define BN 256
#define BK 32
#define LSTR 40  // 32 + 8 bf16 pad => 80B row stride; 16B aligned, conflict-free frags
#define SA_STAGE (BM * LSTR)
#define SB_STAGE (BN * LSTR)

__global__ __launch_bounds__(256)
void wmma_gemm_nt(const __bf16* __restrict__ A,
                  const __bf16* __restrict__ Bt,
                  const float* __restrict__ bias,   // length N, or null
                  float* __restrict__ Cf,           // fp32 out, or null
                  __bf16* __restrict__ Cb,          // bf16 out, or null
                  int M, int N, int K, int doRelu) {
    __shared__ __bf16 sA[2][SA_STAGE];
    __shared__ __bf16 sB[2][SB_STAGE];
    const int tid  = threadIdx.x;
    const int lane = tid & 31;
    const int wave = tid >> 5;
    const int wm = wave & 1;        // 2 waves along M (64 rows each)
    const int wn = wave >> 1;       // 4 waves along N (64 cols each)
    const int tileM = blockIdx.y * BM;
    const int tileN = blockIdx.x * BN;

    // Staging plan per stage: A = 128 rows x 4 quads(16B), B = 256 rows x 4 quads.
    // Thread covers A rows {r0, r0+64}, B rows {r0, r0+64, r0+128, r0+192}.
    const int r0 = tid >> 2;               // 0..63
    const int cq = (tid & 3) << 3;         // element offset 0,8,16,24
    const uint64_t rowBytes = (uint64_t)K * 2;
    uint64_t gA0 = (uint64_t)(uintptr_t)(A  + (size_t)(tileM + r0) * K + cq);
    uint64_t gA1 = gA0 + 64 * rowBytes;
    uint64_t gB0 = (uint64_t)(uintptr_t)(Bt + (size_t)(tileN + r0) * K + cq);
    uint64_t gB1 = gB0 +  64 * rowBytes;
    uint64_t gB2 = gB0 + 128 * rowBytes;
    uint64_t gB3 = gB0 + 192 * rowBytes;
    const uint32_t la0[2] = { lds_lo32(&sA[0][r0 * LSTR + cq]),
                              lds_lo32(&sA[1][r0 * LSTR + cq]) };
    const uint32_t lb0[2] = { lds_lo32(&sB[0][r0 * LSTR + cq]),
                              lds_lo32(&sB[1][r0 * LSTR + cq]) };
    const uint32_t stepLds = 64 * LSTR * 2;   // 64 rows in bytes

    auto issue_stage = [&](int s, int k0) {
        const uint64_t kb = (uint64_t)k0 * 2;
        async_ld_b128(la0[s],               gA0 + kb);
        async_ld_b128(la0[s] + stepLds,     gA1 + kb);
        async_ld_b128(lb0[s],               gB0 + kb);
        async_ld_b128(lb0[s] + stepLds,     gB1 + kb);
        async_ld_b128(lb0[s] + 2 * stepLds, gB2 + kb);
        async_ld_b128(lb0[s] + 3 * stepLds, gB3 + kb);
    };

    const int lrow  = lane & 15;
    const int khalf = (lane >> 4) << 3;    // lanes 16..31 take K offset +8

    v8f acc[4][4] = {};

    issue_stage(0, 0);
    int cur = 0;
    for (int k0 = 0; k0 < K; k0 += BK) {
        wait_async0();          // stage `cur` landed in LDS (wave-local)
        __syncthreads();        // ...and visible to / complete for all waves
        if (k0 + BK < K) issue_stage(cur ^ 1, k0 + BK);

        // Fragments per ISA 16-bit A layout:
        //  lanes 0-15 : row = lane,    K = {0..7, 16..23}
        //  lanes 16-31: row = lane-16, K = {8..15, 24..31}
        union Frag { v16bf v; uint4 q[2]; };
        Frag fa[4], fb[4];
#pragma unroll
        for (int m = 0; m < 4; ++m) {
            const __bf16* p = &sA[cur][(wm * 64 + m * 16 + lrow) * LSTR + khalf];
            fa[m].q[0] = *(const uint4*)p;
            fa[m].q[1] = *(const uint4*)(p + 16);
        }
#pragma unroll
        for (int n = 0; n < 4; ++n) {
            const __bf16* p = &sB[cur][(wn * 64 + n * 16 + lrow) * LSTR + khalf];
            fb[n].q[0] = *(const uint4*)p;
            fb[n].q[1] = *(const uint4*)(p + 16);
        }
#pragma unroll
        for (int m = 0; m < 4; ++m)
#pragma unroll
            for (int n = 0; n < 4; ++n)
                acc[m][n] = __builtin_amdgcn_wmma_f32_16x16x32_bf16(
                    false, fa[m].v, false, fb[n].v,
                    (short)0, acc[m][n], false, false);
        cur ^= 1;
    }

    // Epilogue. C layout: VGPR i -> M = i (lanes<16) / 8+i (lanes>=16), N = lane&15.
    const int rbase = tileM + wm * 64 + ((lane >> 4) << 3);
    const int cbase = tileN + wn * 64 + lrow;
#pragma unroll
    for (int n = 0; n < 4; ++n) {
        const int col = cbase + n * 16;
        const float bv = bias ? bias[col] : 0.f;
#pragma unroll
        for (int m = 0; m < 4; ++m) {
#pragma unroll
            for (int i = 0; i < 8; ++i) {
                float v = acc[m][n][i] + bv;
                if (doRelu) v = fmaxf(v, 0.f);
                const size_t idx = (size_t)(rbase + m * 16 + i) * N + col;
                if (Cf) Cf[idx] = v;
                if (Cb) Cb[idx] = (__bf16)v;
            }
        }
    }
}

// ======================= driver =======================
extern "C" void kernel_launch(void* const* d_in, const int* in_sizes, int n_in,
                              void* d_out, int out_size, void* d_ws, size_t ws_size,
                              hipStream_t stream) {
    (void)in_sizes; (void)n_in; (void)out_size; (void)ws_size;
    const float* x  = (const float*)d_in[0];
    const float* W1 = (const float*)d_in[1];
    const float* b1 = (const float*)d_in[2];
    const float* W2 = (const float*)d_in[3];
    const float* b2 = (const float*)d_in[4];
    const float* W3 = (const float*)d_in[5];
    const float* b3 = (const float*)d_in[6];
    const float* p0 = (const float*)d_in[7];
    const float* p1 = (const float*)d_in[8];

    char* ws = (char*)d_ws;
    const size_t MB = 1ull << 20;
    float*  LT  = (float*)(ws + 0);              // 64MB sinkhorn scratch
    __bf16* P0b = (__bf16*)(ws + 64 * MB);       // 32MB
    __bf16* P1b = (__bf16*)(ws + 96 * MB);       // 32MB
    __bf16* W2t = (__bf16*)(ws + 128 * MB);      // 32MB  W2^T
    __bf16* W1t = (__bf16*)(ws + 160 * MB);      // 8MB   W1^T
    __bf16* W3b = (__bf16*)(ws + 168 * MB);      // 8MB
    __bf16* W1p = (__bf16*)(ws + 176 * MB);      // 8MB
    __bf16* W3p = (__bf16*)(ws + 184 * MB);      // 8MB
    __bf16* xb  = (__bf16*)(ws + 192 * MB);      // 0.5MB
    __bf16* h1  = (__bf16*)(ws + 193 * MB);      // 2MB
    __bf16* h2  = (__bf16*)(ws + 195 * MB);      // 2MB
    float*  b1p = (float*)(ws + 197 * MB);       // 16KB
    float*  b2p = (float*)(ws + 197 * MB + 64 * 1024);
    __bf16* T1  = (__bf16*)(ws + 0);             // alias LT (free after sinkhorn)
    __bf16* W2p = (__bf16*)(ws + 32 * MB);       // alias LT upper half

    // ---- Sinkhorn: p0 -> P0b, p1 -> P1b ----
    const float* ps[2] = {p0, p1};
    __bf16* Pb[2] = {P0b, P1b};
    for (int t = 0; t < 2; ++t) {
        for (int it = 0; it < 20; ++it) {
            sk_row<<<H_DIM, 256, 0, stream>>>(it == 0 ? ps[t] : LT, LT);
            sk_col<<<H_DIM / 32, 256, 0, stream>>>(LT);
        }
        exp_to_bf16<<<(H_DIM * (size_t)H_DIM) / 1024, 256, 0, stream>>>(LT, Pb[t]);
    }

    // ---- operand conversions ----
    transpose_to_bf16<<<dim3(D_IN / 32, H_DIM / 32), 256, 0, stream>>>(W1, W1t, H_DIM, D_IN);
    transpose_to_bf16<<<dim3(H_DIM / 32, H_DIM / 32), 256, 0, stream>>>(W2, W2t, H_DIM, H_DIM);
    cvt_to_bf16<<<((size_t)D_OUT * H_DIM) / 1024, 256, 0, stream>>>(W3, W3b);
    cvt_to_bf16<<<((size_t)BATCH * D_IN) / 1024, 256, 0, stream>>>(x, xb);

    // ---- permuted biases ----
    gemv_bf16<<<H_DIM / 8, 256, 0, stream>>>(P0b, b1, b1p);
    gemv_bf16<<<H_DIM / 8, 256, 0, stream>>>(P1b, b2, b2p);

    // ---- weight remapping (C = A * Bt^T) ----
    wmma_gemm_nt<<<dim3(H_DIM / BN, H_DIM / BM), 256, 0, stream>>>(
        P1b, W2t, nullptr, nullptr, T1, H_DIM, H_DIM, H_DIM, 0);        // T1 = P1 @ W2
    wmma_gemm_nt<<<dim3(H_DIM / BN, H_DIM / BM), 256, 0, stream>>>(
        T1, P0b, nullptr, nullptr, W2p, H_DIM, H_DIM, H_DIM, 0);        // W2p = T1 @ P0^T
    wmma_gemm_nt<<<dim3(D_IN / BN, H_DIM / BM), 256, 0, stream>>>(
        P0b, W1t, nullptr, nullptr, W1p, H_DIM, D_IN, H_DIM, 0);        // W1p = P0 @ W1
    wmma_gemm_nt<<<dim3(H_DIM / BN, D_OUT / BM), 256, 0, stream>>>(
        W3b, P1b, nullptr, nullptr, W3p, D_OUT, H_DIM, H_DIM, 0);       // W3p = W3 @ P1^T

    // ---- MLP forward ----
    wmma_gemm_nt<<<dim3(H_DIM / BN, BATCH / BM), 256, 0, stream>>>(
        xb, W1p, b1p, nullptr, h1, BATCH, H_DIM, D_IN, 1);
    wmma_gemm_nt<<<dim3(H_DIM / BN, BATCH / BM), 256, 0, stream>>>(
        h1, W2p, b2p, nullptr, h2, BATCH, H_DIM, H_DIM, 1);
    wmma_gemm_nt<<<dim3(D_OUT / BN, BATCH / BM), 256, 0, stream>>>(
        h2, W3p, b3, (float*)d_out, nullptr, BATCH, D_OUT, H_DIM, 0);
}
